// SpanMarkerV2_73486890435173
// MI455X (gfx1250) — compile-verified, matrix operand order checked by reading, not run on hardware
//
#include <hip/hip_runtime.h>
#include <hip/hip_bf16.h>
#include <stdint.h>

// ---------------------------------------------------------------------------
// Problem constants (match reference)
// ---------------------------------------------------------------------------
#define BB        4
#define LL        512
#define DD        768
#define NSPANS    4096
#define DFF       3072
#define NROWS     (BB * NSPANS)        // 16384 span rows
#define FCHUNK    128                  // DFF chunk width per outer iteration
#define NCHUNKS   (DFF / FCHUNK)       // 24
#define MTILE     32                   // span rows per workgroup (2 WMMA M-tiles)

typedef __attribute__((ext_vector_type(16))) _Float16 v16h;
typedef __attribute__((ext_vector_type(8)))  float    v8f;

union FragU { uint4 u[2]; v16h v; };
union H4U   { _Float16 h[4]; uint2 u; };

// Load a 16-bit WMMA operand fragment for one lane from a row-major matrix:
// elements j=0..7 at p[0..7], elements j=8..15 at p[16..23]
// (caller passes p = rowBase + k0 + 8*(lane>>4); ISA 7.12.2 16-bit layout)
__device__ __forceinline__ v16h load_frag16(const _Float16* p) {
    FragU f;
    f.u[0] = *reinterpret_cast<const uint4*>(p);
    f.u[1] = *reinterpret_cast<const uint4*>(p + 16);
    return f.v;
}

__device__ __forceinline__ v8f zero_v8f() {
    v8f z;
#pragma unroll
    for (int i = 0; i < 8; ++i) z[i] = 0.0f;
    return z;
}

// ---------------------------------------------------------------------------
// Kernel 0: LDS-tiled transpose + f32->f16 convert.
//   src: [R][C] f32 (row-major)  ->  dst: [C][R] f16 (row-major)
//   grid = (R/32, C/32), 256 threads (32x8)
// ---------------------------------------------------------------------------
__global__ void __launch_bounds__(256)
k_transpose_cvt(const float* __restrict__ src, _Float16* __restrict__ dst,
                int R, int C) {
    __shared__ _Float16 tile[32][34];       // +2 pad: kill bank conflicts
    const int r0 = blockIdx.x * 32;         // row base in src (col base in dst)
    const int c0 = blockIdx.y * 32;         // col base in src (row base in dst)
    const int tx = threadIdx.x & 31;
    const int ty = threadIdx.x >> 5;        // 0..7
#pragma unroll
    for (int r = 0; r < 32; r += 8)         // coalesced read of src
        tile[r + ty][tx] = (_Float16)src[(size_t)(r0 + r + ty) * C + c0 + tx];
    __syncthreads();
#pragma unroll
    for (int r = 0; r < 32; r += 8)         // coalesced write of dst
        dst[(size_t)(c0 + r + ty) * R + r0 + tx] = tile[tx][r + ty];
}

// ---------------------------------------------------------------------------
// Kernel 1: gather + average spans -> A (NROWS x DD, f16). One wave per span.
// ---------------------------------------------------------------------------
__global__ void k_gather_avg(const float* __restrict__ h,
                             const int*   __restrict__ span_idx,
                             _Float16*    __restrict__ A) {
    const int g  = blockIdx.x * 8 + (threadIdx.x >> 5);   // span row 0..NROWS-1
    const int ln = threadIdx.x & 31;
    if (g >= NROWS) return;
    const int b     = g >> 12;                            // / NSPANS
    const int start = span_idx[2 * g + 0];
    const int end   = span_idx[2 * g + 1];
    const int len   = end - start + 1;
    const float inv = 1.0f / (float)len;

    const float* hb = h + ((size_t)b * LL + start) * DD;
    float4 acc[6];
#pragma unroll
    for (int j = 0; j < 6; ++j) { acc[j].x = acc[j].y = acc[j].z = acc[j].w = 0.0f; }

    for (int p = 0; p < len; ++p) {
        const float4* row = reinterpret_cast<const float4*>(hb + (size_t)p * DD);
#pragma unroll
        for (int j = 0; j < 6; ++j) {
            float4 v = row[j * 32 + ln];
            acc[j].x += v.x; acc[j].y += v.y; acc[j].z += v.z; acc[j].w += v.w;
        }
    }
    _Float16* dst = A + (size_t)g * DD;
#pragma unroll
    for (int j = 0; j < 6; ++j) {
        H4U t;
        t.h[0] = (_Float16)(acc[j].x * inv);
        t.h[1] = (_Float16)(acc[j].y * inv);
        t.h[2] = (_Float16)(acc[j].z * inv);
        t.h[3] = (_Float16)(acc[j].w * inv);
        *reinterpret_cast<uint2*>(dst + (size_t)(j * 32 + ln) * 4) = t.u;
    }
}

// ---------------------------------------------------------------------------
// Kernel 2: fused MLP.  One workgroup (8 waves) per 32-span tile.
//   GEMM1: each wave owns one 16-wide DFF column slice per 128-chunk and
//          multiplies each W1 B-fragment against TWO A M-tiles (2 WMMA/load).
//   GEMM2: 12 persistent accumulator tiles per wave (2 M x 6 N = 32x96 f32),
//          each W2 B-fragment reused by 2 WMMAs.
// ---------------------------------------------------------------------------
__global__ void __launch_bounds__(256)
k_fused_mlp(const _Float16* __restrict__ A,
            const _Float16* __restrict__ W1T,     // [DFF][DD]
            const _Float16* __restrict__ W2T,     // [DD][DFF]
            const float*    __restrict__ b1,
            const float*    __restrict__ b2,
            float*          __restrict__ out) {
    const int row0 = blockIdx.x * MTILE;          // span-row base of this tile
    const int tid  = threadIdx.x;
    const int wv   = tid >> 5;                    // wave 0..7
    const int ln   = tid & 31;
    const int lm   = ln & 15;                     // M (or N) index within a tile
    const int kbh  = (ln >> 4) * 8;               // per-lane K sub-offset

    __shared__ __align__(16) _Float16 sA[MTILE][776];   // 49664 B
    __shared__ __align__(16) _Float16 sH[MTILE][136];   //  8704 B

    // ---- stage A tile (32 x 768 f16) into LDS, 12 uint4 per thread ----
    {
        const int r  = tid >> 3;                  // 0..31 (8 threads per row)
        const int c0 = (tid & 7) * 96;            // 96 halves = 12 uint4
        const uint4* src = reinterpret_cast<const uint4*>(A + (size_t)(row0 + r) * DD + c0);
        uint4* dst = reinterpret_cast<uint4*>(&sA[r][c0]);
#pragma unroll
        for (int i = 0; i < 12; ++i) dst[i] = src[i];
    }
    __syncthreads();

    v8f oacc[12];
#pragma unroll
    for (int t = 0; t < 12; ++t) oacc[t] = zero_v8f();

    const _Float16* a0row = &sA[lm][0];
    const _Float16* a1row = &sA[16 + lm][0];
    const _Float16* h0row = &sH[lm][0];
    const _Float16* h1row = &sH[16 + lm][0];

    for (int ch = 0; ch < NCHUNKS; ++ch) {
        // ============ GEMM1: two 16x16 H tiles per wave (shared B) ==========
        const int n0 = ch * FCHUNK + wv * 16;     // DFF column base for this wave
        const _Float16* brow = W1T + (size_t)(n0 + lm) * DD;
        __builtin_prefetch(W1T + (size_t)(n0 + FCHUNK + lm) * DD, 0, 1);

        v8f hacc0 = zero_v8f();
        v8f hacc1 = zero_v8f();
#pragma unroll
        for (int k0 = 0; k0 < DD; k0 += 32) {
            v16h bf  = load_frag16(brow  + k0 + kbh);
            v16h af0 = load_frag16(a0row + k0 + kbh);
            v16h af1 = load_frag16(a1row + k0 + kbh);
            hacc0 = __builtin_amdgcn_wmma_f32_16x16x32_f16(
                false, af0, false, bf, (short)0, hacc0, false, false);
            hacc1 = __builtin_amdgcn_wmma_f32_16x16x32_f16(
                false, af1, false, bf, (short)0, hacc1, false, false);
        }
        // bias + relu -> sH (lane lm = column n; VGPR r = row r + 8*(ln>>4))
        const float bias1 = b1[n0 + lm];
#pragma unroll
        for (int r = 0; r < 8; ++r) {
            float v0 = hacc0[r] + bias1;
            float v1 = hacc1[r] + bias1;
            v0 = v0 > 0.0f ? v0 : 0.0f;
            v1 = v1 > 0.0f ? v1 : 0.0f;
            sH[r + kbh][wv * 16 + lm]      = (_Float16)v0;
            sH[16 + r + kbh][wv * 16 + lm] = (_Float16)v1;
        }
        __syncthreads();

        // ========= GEMM2: 12 persistent OUT tiles per wave (shared B) =======
#pragma unroll
        for (int t = 0; t < 6; ++t) {
            const int ncol = wv * 96 + t * 16 + lm;           // output D column
            const _Float16* b2row = W2T + (size_t)ncol * DFF + ch * FCHUNK;
            v8f acc0 = oacc[2 * t];
            v8f acc1 = oacc[2 * t + 1];
#pragma unroll
            for (int kk = 0; kk < FCHUNK; kk += 32) {
                v16h bf  = load_frag16(b2row + kk + kbh);
                v16h af0 = load_frag16(h0row + kk + kbh);
                v16h af1 = load_frag16(h1row + kk + kbh);
                acc0 = __builtin_amdgcn_wmma_f32_16x16x32_f16(
                    false, af0, false, bf, (short)0, acc0, false, false);
                acc1 = __builtin_amdgcn_wmma_f32_16x16x32_f16(
                    false, af1, false, bf, (short)0, acc1, false, false);
            }
            oacc[2 * t]     = acc0;
            oacc[2 * t + 1] = acc1;
        }
        __syncthreads();   // protect sH before next chunk overwrites it
    }

    // ---- epilogue: + b2, store f32 output ----
#pragma unroll
    for (int t = 0; t < 6; ++t) {
        const int col  = wv * 96 + t * 16 + lm;
        const float bz = b2[col];
#pragma unroll
        for (int r = 0; r < 8; ++r) {
            const int m0 = r + kbh;
            out[(size_t)(row0 + m0) * DD + col]      = oacc[2 * t][r] + bz;
            out[(size_t)(row0 + 16 + m0) * DD + col] = oacc[2 * t + 1][r] + bz;
        }
    }
}

// ---------------------------------------------------------------------------
// Host launcher
// ---------------------------------------------------------------------------
extern "C" void kernel_launch(void* const* d_in, const int* in_sizes, int n_in,
                              void* d_out, int out_size, void* d_ws, size_t ws_size,
                              hipStream_t stream) {
    const float* h        = (const float*)d_in[0];   // [B][L][D]
    const int*   span_idx = (const int*)  d_in[1];   // [B][NSPANS][2]
    const float* W1       = (const float*)d_in[2];   // [D][DFF]
    const float* b1       = (const float*)d_in[3];   // [DFF]
    const float* W2       = (const float*)d_in[4];   // [DFF][D]
    const float* b2       = (const float*)d_in[5];   // [D]
    float*       out      = (float*)d_out;           // [B][NSPANS][D]

    // workspace layout (f16): A [NROWS][D] | W1T [DFF][D] | W2T [D][DFF]
    char* ws = (char*)d_ws;
    _Float16* Ah  = (_Float16*)(ws);
    _Float16* W1T = (_Float16*)(ws + (size_t)NROWS * DD * 2);
    _Float16* W2T = (_Float16*)(ws + (size_t)NROWS * DD * 2 + (size_t)DFF * DD * 2);

    // W1 [DD][DFF] -> W1T [DFF][DD]
    {
        dim3 g(DD / 32, DFF / 32);
        k_transpose_cvt<<<g, 256, 0, stream>>>(W1, W1T, DD, DFF);
    }
    // W2 [DFF][DD] -> W2T [DD][DFF]
    {
        dim3 g(DFF / 32, DD / 32);
        k_transpose_cvt<<<g, 256, 0, stream>>>(W2, W2T, DFF, DD);
    }
    k_gather_avg<<<NROWS / 8, 256, 0, stream>>>(h, span_idx, Ah);
    k_fused_mlp <<<NROWS / MTILE, 256, 0, stream>>>(Ah, W1T, W2T, b1, b2, out);
    (void)in_sizes; (void)n_in; (void)out_size; (void)ws_size;
}
